// SelfSeqAttention_5025111736456
// MI455X (gfx1250) — matrix operand order repaired and checked by reading, not compiled, not run on hardware
//
#include <hip/hip_runtime.h>
#include <hip/hip_bf16.h>
#include <math.h>

typedef __bf16 bf16_t;
typedef __attribute__((ext_vector_type(16))) __bf16 bf16x16;
typedef __attribute__((ext_vector_type(8)))  float  f32x8;

#define DINL __device__ __forceinline__

// ---------------------------------------------------------------------------
// Fragment loaders (wave32 WMMA 16x16x32 bf16 layouts per CDNA5 ISA 7.12.2)
// A (16x32, MxK): lane m(0-15): K {0..7,16..23}; lane m+16: K {8..15,24..31}
//   -> per lane: two 16B chunks at (k0 + lhalf*8) and (k0 + 16 + lhalf*8)
// B (32x16, KxN): lane n(0-15): K {0..7,16..23}; lane n+16: K {8..15,24..31}
//   -> per lane: two 16B chunks at (k0 + lhalf*8) and (k0 + 16 + lhalf*8)
// Work both from global and LDS pointers (LDS derefs lower to ds_load_b128).
// ---------------------------------------------------------------------------
DINL bf16x16 load_a_frag(const bf16_t* p) {
  union { bf16x16 v; uint4 q[2]; } u;
  u.q[0] = *(const uint4*)(p);
  u.q[1] = *(const uint4*)(p + 16);
  return u.v;
}
DINL bf16x16 load_b_frag(const bf16_t* p) {
  union { bf16x16 v; uint4 q[2]; } u;
  u.q[0] = *(const uint4*)(p);
  u.q[1] = *(const uint4*)(p + 16);
  return u.v;
}

// gfx1250 async DMA: copy 16B from global to LDS, tracked by ASYNCcnt.
DINL void async_copy_b128(unsigned lds_off, const bf16_t* gptr) {
  asm volatile("global_load_async_to_lds_b128 %0, %1, off"
               :: "v"(lds_off), "v"(gptr) : "memory");
}

DINL float sigm(float x) { return 1.0f / (1.0f + __expf(-x)); }

// block reduction helper (wave32 shuffle + LDS across up to 8 waves)
DINL float block_reduce(float v, float* red, bool is_max) {
#pragma unroll
  for (int o = 16; o > 0; o >>= 1) {
    float t = __shfl_xor(v, o, 32);
    v = is_max ? fmaxf(v, t) : (v + t);
  }
  const int nw = blockDim.x >> 5;
  if ((threadIdx.x & 31) == 0) red[threadIdx.x >> 5] = v;
  __syncthreads();
  if (threadIdx.x == 0) {
    float r = red[0];
    for (int i = 1; i < nw; ++i) r = is_max ? fmaxf(r, red[i]) : (r + red[i]);
    red[8] = r;
  }
  __syncthreads();
  float r = red[8];
  __syncthreads();
  return r;
}

// atomic grid barrier (bar[0]=counter, bar[1]=generation)
DINL void gridsync(volatile unsigned* bar) {
  __syncthreads();
  if (threadIdx.x == 0) {
    __threadfence();
    unsigned g = bar[1];
    if (atomicAdd((unsigned*)bar, 1u) == gridDim.x - 1) {
      bar[0] = 0;
      __threadfence();
      atomicAdd((unsigned*)(bar + 1), 1u);
    } else {
      while (bar[1] == g) { __builtin_amdgcn_s_sleep(2); }
    }
    __threadfence();
  }
  __syncthreads();
}

// ---------------------------------------------------------------------------
// Generic WMMA GEMM:  C[m,n] = act( A[m,:] . B[n,:] + bias[n] (+ C) )
// A: bf16 [M,K] lda (K contiguous).  B: bf16 [N,K] ldb (K contiguous).
// 8 waves/block; wave tile 32x64. Block tile 128x128, K chunked by 64 and
// double-buffered in LDS via global_load_async_to_lds_b128 (ASYNCcnt), so
// each block reads every A/B element exactly once from L2 (32KB/chunk)
// instead of 96KB with redundant per-wave loads.
// Requires M%128==0, N%128==0, K%64==0.
// flags: 1=accumulate into Cf, 2=LeakyReLU(0.25), 4=ReLU
// z batching: zo=z/zdiv, zi=z%zdiv; offset = zo*s?o + zi*s?i  (elements)
// ---------------------------------------------------------------------------
__global__ __launch_bounds__(256)
void k_gemm_wmma(const bf16_t* __restrict__ A, long sAo, long sAi,
                 const bf16_t* __restrict__ Bw, long sBo, long sBi,
                 const float* __restrict__ bias,
                 float* __restrict__ Cf, bf16_t* __restrict__ Cb,
                 long sCo, long sCi,
                 int K, int lda, int ldb, int ldc, int zdiv, int flags) {
  __shared__ __align__(16) bf16_t sA[2][128 * 64];
  __shared__ __align__(16) bf16_t sB[2][128 * 64];

  const int z  = blockIdx.z;
  const int zo = z / zdiv, zi = z - zo * zdiv;
  A  += (long)zo * sAo + (long)zi * sAi;
  Bw += (long)zo * sBo + (long)zi * sBi;
  const long coff = (long)zo * sCo + (long)zi * sCi;

  const int tid  = threadIdx.x;
  const int wave = tid >> 5;
  const int lane = tid & 31;
  const int lmod = lane & 15, lhalf = lane >> 4;
  const int waveM = wave & 3, waveN = wave >> 2;
  const int row0 = blockIdx.y * 128 + waveM * 32;
  const int col0 = blockIdx.x * 128 + waveN * 64;
  const int rowB = blockIdx.y * 128;             // block A-tile origin
  const int colB = blockIdx.x * 128;             // block B-tile origin

  // staging addressing: 1024 x 16B chunks per tile; 4 per thread.
  // chunk c: row = c>>3 (0..127), sub = c&7 (8 x 16B along 64 K-values)
  const int c0  = tid;                           // + it*256
  const int srow = c0 >> 3, ssub = c0 & 7;

  auto stage = [&](int buf, int kc) {
#pragma unroll
    for (int it = 0; it < 4; ++it) {
      const int row = srow + it * 32, sub = ssub;
      const bf16_t* ga = A  + (long)(rowB + row) * lda + kc + sub * 8;
      const bf16_t* gb = Bw + (long)(colB + row) * ldb + kc + sub * 8;
      async_copy_b128((unsigned)(uintptr_t)(&sA[buf][row * 64 + sub * 8]), ga);
      async_copy_b128((unsigned)(uintptr_t)(&sB[buf][row * 64 + sub * 8]), gb);
    }
  };

  const int nchunk = K >> 6;
  f32x8 acc[2][4] = {};

  stage(0, 0);
  int buf = 0;
  for (int ci = 0; ci < nchunk; ++ci) {
    if (ci + 1 < nchunk) {
      stage(buf ^ 1, (ci + 1) << 6);
      asm volatile("s_wait_asynccnt 0x8" ::: "memory");  // current chunk landed
    } else {
      asm volatile("s_wait_asynccnt 0x0" ::: "memory");
    }
    __syncthreads();

    const bf16_t* la = &sA[buf][(waveM * 32 + lmod) * 64 + lhalf * 8];
    const bf16_t* lb = &sB[buf][(waveN * 64 + lmod) * 64 + lhalf * 8];
#pragma unroll
    for (int ks = 0; ks < 64; ks += 32) {
      bf16x16 a0 = load_a_frag(la + ks);
      bf16x16 a1 = load_a_frag(la + 16 * 64 + ks);
      bf16x16 b0 = load_b_frag(lb + ks);
      bf16x16 b1 = load_b_frag(lb + 16 * 64 + ks);
      bf16x16 b2 = load_b_frag(lb + 32 * 64 + ks);
      bf16x16 b3 = load_b_frag(lb + 48 * 64 + ks);
      acc[0][0] = __builtin_amdgcn_wmma_f32_16x16x32_bf16(false, a0, false, b0, (short)0, acc[0][0], false, false);
      acc[0][1] = __builtin_amdgcn_wmma_f32_16x16x32_bf16(false, a0, false, b1, (short)0, acc[0][1], false, false);
      acc[0][2] = __builtin_amdgcn_wmma_f32_16x16x32_bf16(false, a0, false, b2, (short)0, acc[0][2], false, false);
      acc[0][3] = __builtin_amdgcn_wmma_f32_16x16x32_bf16(false, a0, false, b3, (short)0, acc[0][3], false, false);
      acc[1][0] = __builtin_amdgcn_wmma_f32_16x16x32_bf16(false, a1, false, b0, (short)0, acc[1][0], false, false);
      acc[1][1] = __builtin_amdgcn_wmma_f32_16x16x32_bf16(false, a1, false, b1, (short)0, acc[1][1], false, false);
      acc[1][2] = __builtin_amdgcn_wmma_f32_16x16x32_bf16(false, a1, false, b2, (short)0, acc[1][2], false, false);
      acc[1][3] = __builtin_amdgcn_wmma_f32_16x16x32_bf16(false, a1, false, b3, (short)0, acc[1][3], false, false);
    }
    __syncthreads();
    buf ^= 1;
  }

  // C/D layout: VGPR r -> m = r + (lane>=16 ? 8 : 0), n = lane&15
#pragma unroll
  for (int i = 0; i < 2; ++i) {
#pragma unroll
    for (int j = 0; j < 4; ++j) {
#pragma unroll
      for (int r = 0; r < 8; ++r) {
        const int m = row0 + i * 16 + lhalf * 8 + r;
        const int n = col0 + j * 16 + lmod;
        const long ci = coff + (long)m * ldc + n;
        float v = acc[i][j][r];
        if (bias) v += bias[n];
        if (flags & 1) v += Cf[ci];
        if (flags & 2) v = (v >= 0.0f) ? v : 0.25f * v;
        if (flags & 4) v = fmaxf(v, 0.0f);
        if (Cf) Cf[ci] = v;
        if (Cb) Cb[ci] = (bf16_t)v;
      }
    }
  }
}

// ---------------------------------------------------------------------------
// Persistent LSTM: 32 blocks x 8 waves = 256 waves = 4096/16 gate tiles.
// Per step: wave g computes gh[0:16, g*16:+16] = h16(16x1024 bf16) @ w_hh^T
// via 32 WMMAs (batch padded 8->16, rows 8..15 zero), adds precomputed
// x-gates, grid-syncs, then 8192 threads apply the LSTM cell pointwise.
// Direct global loads: w_hh columns are wave-private (no reuse to stage).
// ---------------------------------------------------------------------------
__global__ __launch_bounds__(256)
void k_lstm(const bf16_t* __restrict__ whh, const float* __restrict__ gx,
            bf16_t* __restrict__ h16, float* __restrict__ cst,
            float* __restrict__ gbuf, float* __restrict__ lout,
            unsigned* __restrict__ bar, int T) {
  const int H = 1024, G = 4096;
  const int wave = threadIdx.x >> 5, lane = threadIdx.x & 31;
  const int lmod = lane & 15, lhalf = lane >> 4;
  const int gw = blockIdx.x * 8 + wave;          // 0..255
  const int n0 = gw * 16;
  const bf16_t* bbase = whh + (long)(n0 + lmod) * H + lhalf * 8;
  const bf16_t* abase = h16 + (long)lmod * H + lhalf * 8;
  const int tid = blockIdx.x * 256 + threadIdx.x; // 0..8191
  const int um = tid >> 10, uj = tid & 1023;

  for (int t = 0; t < T; ++t) {
    f32x8 acc = {};
    for (int k0 = 0; k0 < H; k0 += 32) {
      bf16x16 a = load_a_frag(abase + k0);
      bf16x16 b = load_b_frag(bbase + k0);
      acc = __builtin_amdgcn_wmma_f32_16x16x32_bf16(false, a, false, b, (short)0, acc, false, false);
    }
    if (lhalf == 0) {                             // rows m=0..7 are real batch
#pragma unroll
      for (int r = 0; r < 8; ++r) {
        const int n = n0 + lmod;
        gbuf[r * G + n] = acc[r] + gx[((long)r * T + t) * G + n];
      }
    }
    gridsync(bar);
    {
      float gi = gbuf[um * G + uj];
      float gf = gbuf[um * G + 1024 + uj];
      float gg = gbuf[um * G + 2048 + uj];
      float go = gbuf[um * G + 3072 + uj];
      float c = sigm(gf) * cst[tid] + sigm(gi) * tanhf(gg);
      float h = sigm(go) * tanhf(c);
      cst[tid] = c;
      lout[((long)um * T + t) * H + uj] = h;
      h16[um * H + uj] = (bf16_t)h;
    }
    gridsync(bar);
  }
}

// ---------------------------------------------------------------------------
// Support kernels
// ---------------------------------------------------------------------------
__global__ void k_convert(const float* __restrict__ s, bf16_t* __restrict__ d, long n) {
  long i = (long)blockIdx.x * blockDim.x + threadIdx.x;
  if (i < n) d[i] = (bf16_t)s[i];
}

// w_c1 [512][1024][3] -> dst[k][co][ci] bf16 (3 K-major weight matrices)
__global__ void k_repack_wc1(const float* __restrict__ w, bf16_t* __restrict__ d) {
  long i = (long)blockIdx.x * blockDim.x + threadIdx.x;
  if (i >= 3L * 512 * 1024) return;
  int ci = (int)(i & 1023), co = (int)((i >> 10) & 511), k = (int)(i >> 19);
  d[i] = (bf16_t)w[((long)co * 1024 + ci) * 3 + k];
}

__global__ void k_addbias(const float* __restrict__ a, const float* __restrict__ b,
                          float* __restrict__ d, int n) {
  int i = blockIdx.x * blockDim.x + threadIdx.x;
  if (i < n) d[i] = a[i] + b[i];
}

__global__ void k_init(bf16_t* h16, float* cst, unsigned* bar) {
  int i = blockIdx.x * blockDim.x + threadIdx.x;
  if (i < 16 * 1024) h16[i] = (bf16_t)0.0f;
  if (i < 8192) cst[i] = 0.0f;
  if (i < 2) bar[i] = 0u;
}

// v [8][1024][2048] (b,s,nh*512+d) -> vt [32][512][1024] (z,d,s)
__global__ void k_vt(const bf16_t* __restrict__ v, bf16_t* __restrict__ vt) {
  long i = (long)blockIdx.x * blockDim.x + threadIdx.x;
  if (i >= 8L * 4 * 512 * 1024) return;
  int s = (int)(i & 1023), d = (int)((i >> 10) & 511);
  int z = (int)(i >> 19), b = z >> 2, nh = z & 3;
  vt[i] = v[((long)b * 1024 + s) * 2048 + nh * 512 + d];
}

__global__ void k_softmax_bf16(const float* __restrict__ S, bf16_t* __restrict__ O, int n) {
  __shared__ float red[9];
  const long row = blockIdx.x;
  const float* p = S + row * (long)n;
  bf16_t* o = O + row * (long)n;
  float mx = -3.4e38f;
  for (int i = threadIdx.x; i < n; i += blockDim.x) mx = fmaxf(mx, p[i]);
  mx = block_reduce(mx, red, true);
  float s = 0.0f;
  for (int i = threadIdx.x; i < n; i += blockDim.x) s += __expf(p[i] - mx);
  s = block_reduce(s, red, false);
  float inv = 1.0f / s;
  for (int i = threadIdx.x; i < n; i += blockDim.x) o[i] = (bf16_t)(__expf(p[i] - mx) * inv);
}

__global__ void k_softmax_f32(float* __restrict__ S, int n) {
  __shared__ float red[9];
  float* p = S + blockIdx.x * (long)n;
  float mx = -3.4e38f;
  for (int i = threadIdx.x; i < n; i += blockDim.x) mx = fmaxf(mx, p[i]);
  mx = block_reduce(mx, red, true);
  float s = 0.0f;
  for (int i = threadIdx.x; i < n; i += blockDim.x) s += __expf(p[i] - mx);
  s = block_reduce(s, red, false);
  float inv = 1.0f / s;
  for (int i = threadIdx.x; i < n; i += blockDim.x) p[i] = __expf(p[i] - mx) * inv;
}

__global__ void k_layernorm(const float* __restrict__ Y, const float* __restrict__ g,
                            const float* __restrict__ b, bf16_t* __restrict__ O) {
  __shared__ float red[9];
  const long row = blockIdx.x;
  const float* p = Y + row * 512;
  float s = 0.0f, s2 = 0.0f;
  for (int i = threadIdx.x; i < 512; i += blockDim.x) { float v = p[i]; s += v; s2 += v * v; }
  s  = block_reduce(s,  red, false);
  s2 = block_reduce(s2, red, false);
  float mu = s * (1.0f / 512.0f);
  float var = s2 * (1.0f / 512.0f) - mu * mu;
  float r = rsqrtf(var + 1e-5f);
  for (int i = threadIdx.x; i < 512; i += blockDim.x)
    O[row * 512 + i] = (bf16_t)((p[i] - mu) * r * g[i] + b[i]);
}

// lstm [8][1024][1024] f32 -> pad [8][1026][1024] bf16 (zero edge rows)
__global__ void k_pad_lstm(const float* __restrict__ l, bf16_t* __restrict__ p) {
  long i = (long)blockIdx.x * blockDim.x + threadIdx.x;
  if (i >= 8L * 1026 * 1024) return;
  int h = (int)(i & 1023);
  long r = i >> 10;
  int tt = (int)(r % 1026), b = (int)(r / 1026);
  float v = 0.0f;
  if (tt >= 1 && tt <= 1024) v = l[((long)b * 1024 + (tt - 1)) * 1024 + h];
  p[i] = (bf16_t)v;
}

// a1 [8][1024][512] (relu'd), w [2][512][3] -> a2 [8][2][1024]
__global__ void k_conv2(const float* __restrict__ a1, const float* __restrict__ w,
                        const float* __restrict__ bias, float* __restrict__ a2) {
  int idx = blockIdx.x * blockDim.x + threadIdx.x;
  if (idx >= 8 * 2 * 1024) return;
  int t = idx & 1023, c = (idx >> 10) & 1, b = idx >> 11;
  float s = bias[c];
  for (int k = 0; k < 3; ++k) {
    int tt = t + k - 1;
    if (tt < 0 || tt >= 1024) continue;
    const float* ap = a1 + ((long)b * 1024 + tt) * 512;
    const float* wp = w + ((long)c * 512) * 3 + k;
    for (int ci = 0; ci < 512; ++ci) s += ap[ci] * wp[ci * 3];
  }
  a2[idx] = s;
}

// feat[b][c*1024+h] = sum_t a2[b][c][t] * lstm[b][t][h]
__global__ void k_feat(const float* __restrict__ a2, const float* __restrict__ l,
                       float* __restrict__ out) {
  int idx = blockIdx.x * blockDim.x + threadIdx.x;
  if (idx >= 8 * 2 * 1024) return;
  int h = idx & 1023, c = (idx >> 10) & 1, b = idx >> 11;
  const float* ap = a2 + ((long)b * 2 + c) * 1024;
  const float* lp = l + (long)b * 1024 * 1024 + h;
  float s = 0.0f;
  for (int t = 0; t < 1024; ++t) s += ap[t] * lp[(long)t * 1024];
  out[idx] = s;
}

// ---------------------------------------------------------------------------
// Workspace layout (bytes). Peak ~328 MB with lifetime-based aliasing.
// ---------------------------------------------------------------------------
#define MB_ (1024LL * 1024LL)
static constexpr long OFF_W_IN1 = 0;
static constexpr long OFF_W_IN2 = OFF_W_IN1 + 524288;
static constexpr long OFF_W_Q   = OFF_W_IN2 + 524288;
static constexpr long OFF_W_K   = OFF_W_Q + 1048576;
static constexpr long OFF_W_V   = OFF_W_K + 1048576;
static constexpr long OFF_W_P   = OFF_W_V + 2097152;
static constexpr long OFF_W_IH  = OFF_W_P + 2097152;
static constexpr long OFF_W_HH  = OFF_W_IH + 4194304;
static constexpr long OFF_WC1   = OFF_W_HH + 8388608;
static constexpr long OFF_BIASG = OFF_WC1 + 3145728;
static constexpr long OFF_H16   = OFF_BIASG + 16384;
static constexpr long OFF_CST   = OFF_H16 + 32768;
static constexpr long OFF_GBUF  = OFF_CST + 32768;
static constexpr long OFF_BAR   = OFF_GBUF + 131072;
static constexpr long OFF_ACT   = 24 * MB_;
static constexpr long OFF_XBF     = OFF_ACT;            // 8MB; later h_bf, later y_f32
static constexpr long OFF_H1BF    = OFF_ACT + 8 * MB_;  // 8MB
static constexpr long OFF_HBF     = OFF_XBF;            // alias (x dead)
static constexpr long OFF_YF32    = OFF_ACT;            // alias 16MB (h/h1 dead)
static constexpr long OFF_QBF     = OFF_ACT + 16 * MB_; // 16MB
static constexpr long OFF_KBF     = OFF_ACT + 32 * MB_; // 16MB
static constexpr long OFF_LSTMPAD = OFF_QBF;            // 16.8MB over q+k (dead)
static constexpr long OFF_SCORES  = OFF_ACT + 48 * MB_; // 128MB fp32
static constexpr long OFF_VBF     = OFF_SCORES;         // v bf16 32MB (dead before scores)
static constexpr long OFF_GATESX  = OFF_SCORES;         // gates fp32 (scores dead)
static constexpr long OFF_VT      = OFF_ACT + 176 * MB_; // 32MB
static constexpr long OFF_LSTMOUT = OFF_VT;             // fp32 32MB (vt dead)
static constexpr long OFF_ATTN    = OFF_ACT + 208 * MB_; // 64MB bf16
static constexpr long OFF_A1      = OFF_ATTN;           // fp32 16MB (attn dead)
static constexpr long OFF_A2      = OFF_ATTN + 16 * MB_; // 64KB
static constexpr long OFF_OBF     = OFF_ACT + 272 * MB_; // 32MB
static constexpr long OFF_YBF     = OFF_OBF;            // bf16 8MB (o dead)

extern "C" void kernel_launch(void* const* d_in, const int* in_sizes, int n_in,
                              void* d_out, int out_size, void* d_ws, size_t ws_size,
                              hipStream_t stream) {
  (void)in_sizes; (void)n_in; (void)out_size; (void)ws_size;
  const float* x     = (const float*)d_in[0];
  const float* w_in1 = (const float*)d_in[1];
  const float* b_in1 = (const float*)d_in[2];
  const float* w_in2 = (const float*)d_in[3];
  const float* b_in2 = (const float*)d_in[4];
  const float* w_q   = (const float*)d_in[5];
  const float* b_q   = (const float*)d_in[6];
  const float* w_k   = (const float*)d_in[7];
  const float* b_k   = (const float*)d_in[8];
  const float* w_v   = (const float*)d_in[9];
  const float* b_v   = (const float*)d_in[10];
  const float* w_p   = (const float*)d_in[11];
  const float* b_p   = (const float*)d_in[12];
  const float* ln_g  = (const float*)d_in[13];
  const float* ln_b  = (const float*)d_in[14];
  const float* w_ih  = (const float*)d_in[15];
  const float* w_hh  = (const float*)d_in[16];
  const float* b_ih  = (const float*)d_in[17];
  const float* b_hh  = (const float*)d_in[18];
  const float* w_c1  = (const float*)d_in[19];
  const float* b_c1  = (const float*)d_in[20];
  const float* w_c2  = (const float*)d_in[21];
  const float* b_c2  = (const float*)d_in[22];

  char* ws = (char*)d_ws;
  auto Wb = [&](long off) -> bf16_t* { return (bf16_t*)(ws + off); };
  auto Ff = [&](long off) -> float*  { return (float*)(ws + off); };

  auto cvt = [&](const float* s, long off, long n) {
    k_convert<<<dim3((unsigned)((n + 255) / 256)), dim3(256), 0, stream>>>(s, Wb(off), n);
  };

  // ---- stage 0: weight/activation bf16 staging, bias fuse, LSTM state init
  cvt(x, OFF_XBF, 8192L * 512);
  cvt(w_in1, OFF_W_IN1, 512L * 512);
  cvt(w_in2, OFF_W_IN2, 512L * 512);
  cvt(w_q, OFF_W_Q, 1024L * 512);
  cvt(w_k, OFF_W_K, 1024L * 512);
  cvt(w_v, OFF_W_V, 2048L * 512);
  cvt(w_p, OFF_W_P, 512L * 2048);
  cvt(w_ih, OFF_W_IH, 4096L * 512);
  cvt(w_hh, OFF_W_HH, 4096L * 1024);
  k_repack_wc1<<<dim3((unsigned)((3L * 512 * 1024 + 255) / 256)), dim3(256), 0, stream>>>(w_c1, Wb(OFF_WC1));
  k_addbias<<<dim3(16), dim3(256), 0, stream>>>(b_ih, b_hh, Ff(OFF_BIASG), 4096);
  k_init<<<dim3(64), dim3(256), 0, stream>>>(Wb(OFF_H16), Ff(OFF_CST), (unsigned*)(ws + OFF_BAR));

  auto gemm = [&](const bf16_t* A, long sAo, long sAi,
                  const bf16_t* Bm, long sBo, long sBi,
                  const float* bias, float* Cf, bf16_t* Cb, long sCo, long sCi,
                  int M, int N, int K, int lda, int ldb, int ldc,
                  int nz, int zdiv, int flags) {
    dim3 g((unsigned)(N / 128), (unsigned)(M / 128), (unsigned)nz);
    k_gemm_wmma<<<g, dim3(256), 0, stream>>>(A, sAo, sAi, Bm, sBo, sBi, bias, Cf, Cb,
                                             sCo, sCi, K, lda, ldb, ldc, zdiv, flags);
  };

  const long TT = 1024L * 1024;

  // ---- stage 1: in_pro (linear -> leaky 0.25 -> linear)
  gemm(Wb(OFF_XBF), 0, 0, Wb(OFF_W_IN1), 0, 0, b_in1, nullptr, Wb(OFF_H1BF), 0, 0,
       8192, 512, 512, 512, 512, 512, 1, 1, /*leaky*/2);
  gemm(Wb(OFF_H1BF), 0, 0, Wb(OFF_W_IN2), 0, 0, b_in2, nullptr, Wb(OFF_HBF), 0, 0,
       8192, 512, 512, 512, 512, 512, 1, 1, 0);

  // ---- stage 2: attention
  gemm(Wb(OFF_HBF), 0, 0, Wb(OFF_W_Q), 0, 0, b_q, nullptr, Wb(OFF_QBF), 0, 0,
       8192, 1024, 512, 512, 512, 1024, 1, 1, 0);
  gemm(Wb(OFF_HBF), 0, 0, Wb(OFF_W_K), 0, 0, b_k, nullptr, Wb(OFF_KBF), 0, 0,
       8192, 1024, 512, 512, 512, 1024, 1, 1, 0);
  gemm(Wb(OFF_HBF), 0, 0, Wb(OFF_W_V), 0, 0, b_v, nullptr, Wb(OFF_VBF), 0, 0,
       8192, 2048, 512, 512, 512, 2048, 1, 1, 0);
  k_vt<<<dim3(65536), dim3(256), 0, stream>>>(Wb(OFF_VBF), Wb(OFF_VT));
  // scores[b,nh] = Q[b,:,nh,:] @ K[b,:,nh,:]^T   (z = b*4+nh)
  gemm(Wb(OFF_QBF), 1048576, 256, Wb(OFF_KBF), 1048576, 256, nullptr,
       Ff(OFF_SCORES), nullptr, 4 * TT, TT,
       1024, 1024, 256, 1024, 1024, 1024, 32, 4, 0);
  k_softmax_bf16<<<dim3(32768), dim3(256), 0, stream>>>(Ff(OFF_SCORES), Wb(OFF_ATTN), 1024);
  // o[b,:,nh*512:+512] = attn[b,nh] @ V_t[b,nh]^T
  gemm(Wb(OFF_ATTN), 4 * TT, TT, Wb(OFF_VT), 4L * 512 * 1024, 512L * 1024, nullptr,
       nullptr, Wb(OFF_OBF), 1024L * 2048, 512,
       1024, 512, 1024, 1024, 1024, 2048, 32, 4, 0);

  // ---- stage 3: projection + LayerNorm
  gemm(Wb(OFF_OBF), 0, 0, Wb(OFF_W_P), 0, 0, b_p, Ff(OFF_YF32), nullptr, 0, 0,
       8192, 512, 2048, 2048, 2048, 512, 1, 1, 0);
  k_layernorm<<<dim3(8192), dim3(256), 0, stream>>>(Ff(OFF_YF32), ln_g, ln_b, Wb(OFF_YBF));

  // ---- stage 4: LSTM (x-gates as one big GEMM, recurrence persistent)
  gemm(Wb(OFF_YBF), 0, 0, Wb(OFF_W_IH), 0, 0, Ff(OFF_BIASG), Ff(OFF_GATESX), nullptr, 0, 0,
       8192, 4096, 512, 512, 512, 4096, 1, 1, 0);
  k_lstm<<<dim3(32), dim3(256), 0, stream>>>(Wb(OFF_W_HH), Ff(OFF_GATESX), Wb(OFF_H16),
                                             Ff(OFF_CST), Ff(OFF_GBUF), Ff(OFF_LSTMOUT),
                                             (unsigned*)(ws + OFF_BAR), 1024);

  // ---- stage 5: conv attention pooling
  k_pad_lstm<<<dim3((unsigned)((8L * 1026 * 1024 + 255) / 256)), dim3(256), 0, stream>>>(
      Ff(OFF_LSTMOUT), Wb(OFF_LSTMPAD));
  for (int k = 0; k < 3; ++k) {   // conv1 as 3 accumulating GEMMs; ReLU on last
    gemm(Wb(OFF_LSTMPAD) + (long)k * 1024, 1050624, 0,
         Wb(OFF_WC1) + (long)k * 524288, 0, 0,
         (k == 0) ? b_c1 : nullptr, Ff(OFF_A1), nullptr, 524288, 0,
         1024, 512, 1024, 1024, 1024, 512, 8, 1, (k == 0) ? 0 : ((k == 2) ? 5 : 1));
  }
  k_conv2<<<dim3(64), dim3(256), 0, stream>>>(Ff(OFF_A1), w_c2, b_c2, Ff(OFF_A2));
  k_softmax_f32<<<dim3(16), dim3(256), 0, stream>>>(Ff(OFF_A2), 1024);
  k_feat<<<dim3(64), dim3(256), 0, stream>>>(Ff(OFF_A2), Ff(OFF_LSTMOUT), (float*)d_out);
}
